// Model_on_GCN_41420664602962
// MI455X (gfx1250) — matrix operand order repaired and checked by reading, not compiled, not run on hardware
//
#include <hip/hip_runtime.h>
#include <hip/hip_bf16.h>

typedef __attribute__((ext_vector_type(16))) _Float16 v16h;
typedef __attribute__((ext_vector_type(8)))  _Float16 v8h;
typedef __attribute__((ext_vector_type(4)))  _Float16 v4h;
typedef __attribute__((ext_vector_type(8)))  float    v8f;

static constexpr int HDIM  = 256;
static constexpr int NATOM = 5000;
static constexpr int NEDGE = 160000;
static constexpr int LDSW  = 40;   // padded LDS row stride in halves (80B: conflict-free, 16B aligned)

enum { EPI_BIAS = 0, EPI_BIAS_RELU = 1, EPI_GCN = 2 };

static __device__ __forceinline__ v16h cat16(v8h lo, v8h hi) {
  return __builtin_shufflevector(lo, hi, 0,1,2,3,4,5,6,7,8,9,10,11,12,13,14,15);
}

// Per-thread staging registers: one 128x32 A quad-column + one 32x128 B quad.
struct Stage {
  float4 a[4];
  float4 b[4];
};

// ---------------------------------------------------------------------------
// Generic tiled GEMM: C[M,N] = epilogue(A[M,K] @ B[K,N]).
// fp32 in memory (K, N multiples of 4 -> aligned float4 quads), f16 in LDS,
// f32 accumulate via v_wmma_f32_16x16x32_f16.
// Block = 256 threads (8 waves), tile 128x128x32; wave w owns rows (w&3)*32,
// cols (w>>2)*64 -> 2x4 fragments.
// Software-pipelined: global loads for tile k+1 are issued before the WMMA
// block of tile k, so their single loadcnt wait lands after the compute.
// ---------------------------------------------------------------------------
template <int EPI>
__global__ __launch_bounds__(256)
void gemm_f16_wmma(const float* __restrict__ A, const float* __restrict__ B,
                   const float* __restrict__ bias, float* __restrict__ C,
                   int M, int N, int K,
                   const float* __restrict__ rowscale,   // EPI_GCN: norm_r[M]
                   const float* __restrict__ residual)   // EPI_GCN: x[M,N]
{
  __shared__ _Float16 As[128 * LDSW];
  __shared__ _Float16 Bs[128 * LDSW];

  const int tid  = threadIdx.x;
  const int lane = tid & 31;
  const int wv   = tid >> 5;
  const int hlf  = lane >> 4;     // half-lane group (K-split per ISA layout)
  const int l16  = lane & 15;
  const int wm   = (wv & 3) * 32;
  const int wn   = (wv >> 2) * 64;
  const long brow = (long)blockIdx.y * 128;
  const long bcol = (long)blockIdx.x * 128;

  const int ar  = tid >> 3;        // A staging: row 0..31 (x4 iters)
  const int ac  = (tid & 7) * 4;   // A staging: aligned col quad
  const int bn4 = (tid & 31) * 4;  // B staging: aligned col quad
  const int bk  = tid >> 5;        // B staging: k 0..7 (x4 iters)

  v8f acc[2][4] = {};
  Stage st;

  // raw loads only: clamped addresses, no use of the data (keeps loads batched)
  auto stage_load = [&](int k0) {
    #pragma unroll
    for (int i = 0; i < 4; ++i) {
      const long grow = brow + ar + 32 * i;
      const int  gk   = k0 + ac;
      const bool ok   = (grow < M) && (gk < K);
      st.a[i] = *(const float4*)(A + (ok ? grow * (long)K + gk : 0));
    }
    #pragma unroll
    for (int i = 0; i < 4; ++i) {
      const int  gk = k0 + bk + 8 * i;
      const long gn = bcol + bn4;
      const bool ok = (gk < K) && (gn < N);
      st.b[i] = *(const float4*)(B + (ok ? (long)gk * N + gn : 0));
    }
  };

  // consume staged registers: zero-select, pack-convert, write LDS
  auto stage_write = [&](int k0) {
    #pragma unroll
    for (int i = 0; i < 4; ++i) {
      const int  row  = ar + 32 * i;
      const long grow = brow + row;
      const int  gk   = k0 + ac;
      const bool ok   = (grow < M) && (gk < K);
      const float4 t  = st.a[i];
      const auto p0 = __builtin_amdgcn_cvt_pkrtz(ok ? t.x : 0.0f, ok ? t.y : 0.0f);
      const auto p1 = __builtin_amdgcn_cvt_pkrtz(ok ? t.z : 0.0f, ok ? t.w : 0.0f);
      const auto q  = __builtin_shufflevector(p0, p1, 0, 1, 2, 3);
      *(v4h*)&As[row * LDSW + ac] = __builtin_bit_cast(v4h, q);
    }
    #pragma unroll
    for (int i = 0; i < 4; ++i) {
      const int  kk = bk + 8 * i;
      const int  gk = k0 + kk;
      const long gn = bcol + bn4;
      const bool ok = (gk < K) && (gn < N);
      const float4 t = st.b[i];
      const auto p0 = __builtin_amdgcn_cvt_pkrtz(ok ? t.x : 0.0f, ok ? t.y : 0.0f);
      const auto p1 = __builtin_amdgcn_cvt_pkrtz(ok ? t.z : 0.0f, ok ? t.w : 0.0f);
      Bs[(bn4 + 0) * LDSW + kk] = (_Float16)p0[0];
      Bs[(bn4 + 1) * LDSW + kk] = (_Float16)p0[1];
      Bs[(bn4 + 2) * LDSW + kk] = (_Float16)p1[0];
      Bs[(bn4 + 3) * LDSW + kk] = (_Float16)p1[1];
    }
  };

  stage_load(0);
  for (int k0 = 0; k0 < K; k0 += 32) {
    stage_write(k0);
    __syncthreads();
    if (k0 + 32 < K) stage_load(k0 + 32);   // uniform branch; overlaps WMMA below

    // ---- fragment loads per ISA 7.12.2 16-bit layouts ----
    v16h af[2], bf[4];
    #pragma unroll
    for (int i = 0; i < 2; ++i) {
      const _Float16* p = &As[(wm + i * 16 + l16) * LDSW];
      af[i] = cat16(*(const v8h*)(p + hlf * 8), *(const v8h*)(p + 16 + hlf * 8));
    }
    #pragma unroll
    for (int j = 0; j < 4; ++j) {
      const _Float16* p = &Bs[(wn + j * 16 + l16) * LDSW];
      bf[j] = cat16(*(const v8h*)(p + hlf * 8), *(const v8h*)(p + 16 + hlf * 8));
    }
    #pragma unroll
    for (int i = 0; i < 2; ++i)
      #pragma unroll
      for (int j = 0; j < 4; ++j)
        acc[i][j] = __builtin_amdgcn_wmma_f32_16x16x32_f16(
            false, af[i], false, bf[j], (short)0, acc[i][j], false, false);
    __syncthreads();
  }

  // ---- epilogue: lane l holds C rows (v + 8*hlf), col l16 per 16x16 tile ----
  const bool full = (brow + 128 <= (long)M) && (bcol + 128 <= (long)N);
  if (full) {
    #pragma unroll
    for (int i = 0; i < 2; ++i) {
      #pragma unroll
      for (int j = 0; j < 4; ++j) {
        const long  n  = bcol + wn + j * 16 + l16;
        const float bv = bias[n];
        #pragma unroll
        for (int v = 0; v < 8; ++v) {
          const long m   = brow + wm + i * 16 + hlf * 8 + v;
          const long idx = m * (long)N + n;
          float val = acc[i][j][v];
          if (EPI == EPI_GCN) {
            val = residual[idx] + val * rowscale[m] + bv;
          } else {
            val += bv;
            if (EPI == EPI_BIAS_RELU) val = fmaxf(val, 0.0f);
          }
          C[idx] = val;
        }
      }
    }
  } else {
    #pragma unroll
    for (int i = 0; i < 2; ++i) {
      #pragma unroll
      for (int j = 0; j < 4; ++j) {
        const long n = bcol + wn + j * 16 + l16;
        #pragma unroll
        for (int v = 0; v < 8; ++v) {
          const long m = brow + wm + i * 16 + hlf * 8 + v;
          if (m < M && n < N) {
            const long idx = m * (long)N + n;
            float val = acc[i][j][v];
            if (EPI == EPI_GCN) {
              val = residual[idx] + val * rowscale[m] + bias[n];
            } else {
              val += bias[n];
              if (EPI == EPI_BIAS_RELU) val = fmaxf(val, 0.0f);
            }
            C[idx] = val;
          }
        }
      }
    }
  }
}

// ---------------------------------------------------------------------------
// BatchNorm batch statistics: 32 columns per block (coalesced reads),
// 8 row-strided partials per column, deterministic tree reduce.
// ---------------------------------------------------------------------------
__global__ __launch_bounds__(256)
void bn_stats(const float* __restrict__ h, int M, int C,
              float* __restrict__ mu, float* __restrict__ var) {
  __shared__ float s1[8][32], s2[8][32];
  const int tx = threadIdx.x & 31;
  const int ty = threadIdx.x >> 5;
  const int c  = blockIdx.x * 32 + tx;
  float sum = 0.0f, sq = 0.0f;
  for (int m = ty; m < M; m += 8) {
    const float v = h[(long)m * C + c];
    sum += v; sq += v * v;
  }
  s1[ty][tx] = sum; s2[ty][tx] = sq;
  __syncthreads();
  for (int s = 4; s > 0; s >>= 1) {
    if (ty < s) { s1[ty][tx] += s1[ty + s][tx];
                  s2[ty][tx] += s2[ty + s][tx]; }
    __syncthreads();
  }
  if (ty == 0) {
    const float m_ = s1[0][tx] / (float)M;
    mu[c]  = m_;
    var[c] = s2[0][tx] / (float)M - m_ * m_;
  }
}

__global__ __launch_bounds__(256)
void bn_relu_apply(float* __restrict__ h, long total, int C,
                   const float* __restrict__ mu, const float* __restrict__ var,
                   const float* __restrict__ g, const float* __restrict__ beta) {
  const long i = (long)blockIdx.x * 256 + threadIdx.x;
  if (i < total) {
    const int c = (int)(i % C);
    const float v = g[c] * (h[i] - mu[c]) * rsqrtf(var[c] + 1e-5f) + beta[c];
    h[i] = fmaxf(v, 0.0f);
  }
}

// ---------------------------------------------------------------------------
// RBF expansion: one thread per (edge, bin) output element (coalesced writes)
// rbf[e,i] = exp(-gamma*(|r_e| - c_i)^2), c_i = 8*i/79, gamma = 79/8
// ---------------------------------------------------------------------------
__global__ __launch_bounds__(256)
void rbf_kernel(const float* __restrict__ r, float* __restrict__ rbf, long total) {
  const long idx = (long)blockIdx.x * 256 + threadIdx.x;
  if (idx < total) {
    const int e = (int)(idx / 80);
    const int i = (int)(idx - (long)e * 80);
    const float x0 = r[3 * e], x1 = r[3 * e + 1], x2 = r[3 * e + 2];
    const float b = sqrtf(x0 * x0 + x1 * x1 + x2 * x2);
    const float d = b - (8.0f * (float)i / 79.0f);
    rbf[idx] = __expf(-9.875f * d * d);
  }
}

// ---------------------------------------------------------------------------
// Degrees + symmetric norms
// ---------------------------------------------------------------------------
__global__ void zero_i32(int* p, int n) {
  const int i = blockIdx.x * 256 + threadIdx.x;
  if (i < n) p[i] = 0;
}
__global__ void zero_f32(float* p, long n) {
  const long i = (long)blockIdx.x * 256 + threadIdx.x;
  if (i < n) p[i] = 0.0f;
}
__global__ void count_deg(const int* __restrict__ src, const int* __restrict__ dst,
                          int* __restrict__ dout, int* __restrict__ din, int E) {
  const int e = blockIdx.x * 256 + threadIdx.x;
  if (e < E) {
    atomicAdd(&dout[src[e]], 1);
    atomicAdd(&din[dst[e]], 1);
  }
}
__global__ void make_norm(const int* __restrict__ dout, const int* __restrict__ din,
                          float* __restrict__ nl, float* __restrict__ nr, int N) {
  const int i = blockIdx.x * 256 + threadIdx.x;
  if (i < N) {
    nl[i] = rsqrtf((float)max(dout[i], 1));
    nr[i] = rsqrtf((float)max(din[i], 1));
  }
}

// ---------------------------------------------------------------------------
// Edge message: sigma = sigmoid(e_src[s]+e_dst[d]+e_edge[e]);
// agg[d] += x[s]*norm_l[s]*sigma.  One block per edge, 256 channels.
// ---------------------------------------------------------------------------
__global__ __launch_bounds__(256)
void edge_scatter(const float* __restrict__ es, const float* __restrict__ ed,
                  const float* __restrict__ ee, const float* __restrict__ x,
                  const float* __restrict__ norm_l,
                  const int* __restrict__ src, const int* __restrict__ dst,
                  float* __restrict__ agg) {
  const int e = blockIdx.x;
  const int c = threadIdx.x;
  const int s = src[e], d = dst[e];
  const float gsum = es[(long)s * HDIM + c] + ed[(long)d * HDIM + c]
                   + ee[(long)e * HDIM + c];
  const float sig = 1.0f / (1.0f + __expf(-gsum));
  const float val = x[(long)s * HDIM + c] * norm_l[s] * sig;
  atomicAdd(&agg[(long)d * HDIM + c], val);
}

// ---------------------------------------------------------------------------
extern "C" void kernel_launch(void* const* d_in, const int* in_sizes, int n_in,
                              void* d_out, int out_size, void* d_ws, size_t ws_size,
                              hipStream_t stream) {
  const float* atom  = (const float*)d_in[0];
  const float* r     = (const float*)d_in[1];
  const int*   src   = (const int*)  d_in[2];
  const int*   dst   = (const int*)  d_in[3];
  const float* ae_w  = (const float*)d_in[4];
  const float* ae_b  = (const float*)d_in[5];
  const float* ae_g  = (const float*)d_in[6];
  const float* ae_bt = (const float*)d_in[7];
  const float* ee1_w = (const float*)d_in[8];
  const float* ee1_b = (const float*)d_in[9];
  const float* ee1_g = (const float*)d_in[10];
  const float* ee1_bt= (const float*)d_in[11];
  const float* ee2_w = (const float*)d_in[12];
  const float* ee2_b = (const float*)d_in[13];
  const float* ee2_g = (const float*)d_in[14];
  const float* ee2_bt= (const float*)d_in[15];
  const float* gcn_w = (const float*)d_in[16];
  const float* gcn_b = (const float*)d_in[17];
  const float* sg_w  = (const float*)d_in[18];
  const float* sg_b  = (const float*)d_in[19];
  const float* dg_w  = (const float*)d_in[20];
  const float* dg_b  = (const float*)d_in[21];
  const float* eg_w  = (const float*)d_in[22];
  const float* eg_b  = (const float*)d_in[23];
  const float* fc1_w = (const float*)d_in[24];
  const float* fc1_b = (const float*)d_in[25];
  const float* fc2_w = (const float*)d_in[26];
  const float* fc2_b = (const float*)d_in[27];
  const float* fc3_w = (const float*)d_in[28];
  const float* fc3_b = (const float*)d_in[29];
  float* out = (float*)d_out;

  // ---- workspace carve-up ----
  float* ws = (float*)d_ws;
  size_t off = 0;
  auto alloc = [&](size_t n) { float* p = ws + off; off += n; return p; };
  float* x     = alloc((size_t)NATOM * HDIM);
  float* e_src = alloc((size_t)NATOM * HDIM);
  float* e_dst = alloc((size_t)NATOM * HDIM);
  float* agg   = alloc((size_t)NATOM * HDIM);
  float* y     = alloc((size_t)NEDGE * HDIM);
  float* eedge = alloc((size_t)NEDGE * HDIM);   // live only in GCN loop
  float* h1    = alloc((size_t)NATOM * 1024);
  float* h2    = alloc((size_t)NATOM * 2048);
  float* mu    = alloc(256);
  float* var_  = alloc(256);
  float* norml = alloc(NATOM);
  float* normr = alloc(NATOM);
  int*   dego  = (int*)alloc(NATOM);
  int*   degi  = (int*)alloc(NATOM);
  // alias pre-loop temporaries into the (not-yet-live) eedge region
  float* rbf   = eedge;                          // NEDGE*80
  float* t64   = eedge + (size_t)NEDGE * 80;     // NEDGE*64

  auto g2 = [](int m, int n) {
    return dim3((unsigned)((n + 127) / 128), (unsigned)((m + 127) / 128));
  };
  auto g1 = [](long n) { return dim3((unsigned)((n + 255) / 256)); };

  // ---- atom embedding MLP: x = relu(BN(atom @ ae_w + ae_b)) ----
  gemm_f16_wmma<EPI_BIAS><<<g2(NATOM, HDIM), 256, 0, stream>>>(
      atom, ae_w, ae_b, x, NATOM, HDIM, 92, nullptr, nullptr);
  bn_stats<<<HDIM / 32, 256, 0, stream>>>(x, NATOM, HDIM, mu, var_);
  bn_relu_apply<<<g1((long)NATOM * HDIM), 256, 0, stream>>>(
      x, (long)NATOM * HDIM, HDIM, mu, var_, ae_g, ae_bt);

  // ---- edge embedding: rbf -> MLP(80->64) -> MLP(64->256) ----
  rbf_kernel<<<g1((long)NEDGE * 80), 256, 0, stream>>>(r, rbf, (long)NEDGE * 80);
  gemm_f16_wmma<EPI_BIAS><<<g2(NEDGE, 64), 256, 0, stream>>>(
      rbf, ee1_w, ee1_b, t64, NEDGE, 64, 80, nullptr, nullptr);
  bn_stats<<<64 / 32, 256, 0, stream>>>(t64, NEDGE, 64, mu, var_);
  bn_relu_apply<<<g1((long)NEDGE * 64), 256, 0, stream>>>(
      t64, (long)NEDGE * 64, 64, mu, var_, ee1_g, ee1_bt);
  gemm_f16_wmma<EPI_BIAS><<<g2(NEDGE, HDIM), 256, 0, stream>>>(
      t64, ee2_w, ee2_b, y, NEDGE, HDIM, 64, nullptr, nullptr);
  bn_stats<<<HDIM / 32, 256, 0, stream>>>(y, NEDGE, HDIM, mu, var_);
  bn_relu_apply<<<g1((long)NEDGE * HDIM), 256, 0, stream>>>(
      y, (long)NEDGE * HDIM, HDIM, mu, var_, ee2_g, ee2_bt);

  // ---- degree norms ----
  zero_i32<<<g1(NATOM), 256, 0, stream>>>(dego, NATOM);
  zero_i32<<<g1(NATOM), 256, 0, stream>>>(degi, NATOM);
  count_deg<<<g1(NEDGE), 256, 0, stream>>>(src, dst, dego, degi, NEDGE);
  make_norm<<<g1(NATOM), 256, 0, stream>>>(dego, degi, norml, normr, NATOM);

  // ---- 4 gated-GCN layers ----
  for (int i = 0; i < 4; ++i) {
    const size_t wo = (size_t)i * HDIM * HDIM;
    const size_t bo = (size_t)i * HDIM;
    gemm_f16_wmma<EPI_BIAS><<<g2(NATOM, HDIM), 256, 0, stream>>>(
        x, sg_w + wo, sg_b + bo, e_src, NATOM, HDIM, HDIM, nullptr, nullptr);
    gemm_f16_wmma<EPI_BIAS><<<g2(NATOM, HDIM), 256, 0, stream>>>(
        x, dg_w + wo, dg_b + bo, e_dst, NATOM, HDIM, HDIM, nullptr, nullptr);
    gemm_f16_wmma<EPI_BIAS><<<g2(NEDGE, HDIM), 256, 0, stream>>>(
        y, eg_w + wo, eg_b + bo, eedge, NEDGE, HDIM, HDIM, nullptr, nullptr);
    zero_f32<<<g1((long)NATOM * HDIM), 256, 0, stream>>>(agg, (long)NATOM * HDIM);
    edge_scatter<<<NEDGE, 256, 0, stream>>>(e_src, e_dst, eedge, x, norml,
                                            src, dst, agg);
    // x += (agg @ gcn_w) * norm_r + gcn_b   (residual fused, in place)
    gemm_f16_wmma<EPI_GCN><<<g2(NATOM, HDIM), 256, 0, stream>>>(
        agg, gcn_w + wo, gcn_b + bo, x, NATOM, HDIM, HDIM, normr, x);
  }

  // ---- readout MLPs ----
  gemm_f16_wmma<EPI_BIAS_RELU><<<g2(NATOM, 1024), 256, 0, stream>>>(
      x, fc1_w, fc1_b, h1, NATOM, 1024, HDIM, nullptr, nullptr);
  gemm_f16_wmma<EPI_BIAS_RELU><<<g2(NATOM, 2048), 256, 0, stream>>>(
      h1, fc2_w, fc2_b, h2, NATOM, 2048, 1024, nullptr, nullptr);
  gemm_f16_wmma<EPI_BIAS_RELU><<<g2(NATOM, 10000), 256, 0, stream>>>(
      h2, fc3_w, fc3_b, out, NATOM, 10000, 2048, nullptr, nullptr);

  (void)in_sizes; (void)n_in; (void)out_size; (void)ws_size;
}
